// MultiHeadAttention_27685359190775
// MI455X (gfx1250) — matrix operand order repaired and checked by reading, not compiled, not run on hardware
//
#include <hip/hip_runtime.h>
#include <math.h>

#define B_ 4
#define S_ 2048
#define D_ 1024
#define H_ 16
#define DH_ 64
#define M_ (B_*S_)

typedef __bf16 bf16;
typedef __attribute__((ext_vector_type(8)))  __bf16 v8bf;
typedef __attribute__((ext_vector_type(16))) __bf16 v16bf;
typedef __attribute__((ext_vector_type(8)))  float  v8f;
typedef int v4i __attribute__((vector_size(16)));

#if __has_builtin(__builtin_amdgcn_global_load_async_to_lds_b128) && \
    __has_builtin(__builtin_amdgcn_s_wait_asynccnt)
#define HAVE_ASYNC_LDS 1
typedef __attribute__((address_space(1))) v4i* gv4i_p;
typedef __attribute__((address_space(3))) v4i* lv4i_p;
#define ASYNC_CP16(gsrc, ldst) \
  __builtin_amdgcn_global_load_async_to_lds_b128((gv4i_p)(gsrc), (lv4i_p)(ldst), 0, 0)
#else
#define HAVE_ASYNC_LDS 0
#endif

static __device__ __forceinline__ v16bf combine16(v8bf lo, v8bf hi) {
  return __builtin_shufflevector(lo, hi, 0,1,2,3,4,5,6,7,8,9,10,11,12,13,14,15);
}

// A-fragment (16x32 bf16). Lane holds row m; half h = lane>>4:
// VGPR0-3 = K[h*8 .. h*8+7], VGPR4-7 = K[16+h*8 .. 16+h*8+7]  (ISA 7.12.2)
static __device__ __forceinline__ v16bf load_a_frag(const bf16* row, int h) {
  v8bf lo = *reinterpret_cast<const v8bf*>(row + h * 8);
  v8bf hi = *reinterpret_cast<const v8bf*>(row + 16 + h * 8);
  return combine16(lo, hi);
}

// B-fragment (32x16 bf16). Lane holds col n = lane&15; half h holds K[h*8 .. h*8+7], K[16+h*8 .. 16+h*8+7]
static __device__ __forceinline__ v16bf load_b_frag(const bf16* row, int h) {
  v8bf lo = *reinterpret_cast<const v8bf*>(row + h * 8);
  v8bf hi = *reinterpret_cast<const v8bf*>(row + 16 + h * 8);
  return combine16(lo, hi);
}

static __device__ __forceinline__ v8f wmma_bf16(v16bf a, v16bf b, v8f c) {
  return __builtin_amdgcn_wmma_f32_16x16x32_bf16(false, a, false, b, (short)0, c,
                                                 false, false);
}

#define V8F_ZERO {0.f,0.f,0.f,0.f,0.f,0.f,0.f,0.f}

// ---------------------------------------------------------------------------
// Kernel 1: QKV projection. C[m,n] = sum_k X[m,k] * W[n,k] + b[n]
// blockIdx.z selects (Wq,bq,Q) / (Wk,bk,K) / (Wv,bv,V). Output bf16 [B,H,S,DH].
// Block tile 128(M) x 128(N), 8 waves; each wave computes 32x64 (8 WMMA/K-step).
// ---------------------------------------------------------------------------
__global__ __launch_bounds__(256) void k_qkv_proj(
    const float* __restrict__ xq, const float* __restrict__ xkv,
    const float* __restrict__ Wq, const float* __restrict__ bq,
    const float* __restrict__ Wk, const float* __restrict__ bk,
    const float* __restrict__ Wv, const float* __restrict__ bv,
    bf16* __restrict__ Qw, bf16* __restrict__ Kw, bf16* __restrict__ Vw)
{
  __shared__ __align__(16) bf16 sA[128][48];
  __shared__ __align__(16) bf16 sB[128][48];

  const float* A; const float* W; const float* bias; bf16* out;
  if (blockIdx.z == 0)      { A = xq;  W = Wq; bias = bq; out = Qw; }
  else if (blockIdx.z == 1) { A = xkv; W = Wk; bias = bk; out = Kw; }
  else                      { A = xkv; W = Wv; bias = bv; out = Vw; }

  const int t = threadIdx.x;
  const int wave = t >> 5, lane = t & 31;
  const int lh = lane >> 4, l16 = lane & 15;
  const int mw = wave & 3, nw = wave >> 2;   // 4 m-blocks of 32, 2 n-blocks of 64
  const int m0 = blockIdx.x * 128;
  const int n0 = blockIdx.y * 128;

  v8f acc[2][4] = { { V8F_ZERO, V8F_ZERO, V8F_ZERO, V8F_ZERO },
                    { V8F_ZERO, V8F_ZERO, V8F_ZERO, V8F_ZERO } };

  for (int k0 = 0; k0 < D_; k0 += 32) {
    // stage A tile 128x32 fp32 -> bf16 (4 float4 per thread)
    #pragma unroll
    for (int i = 0; i < 4; ++i) {
      int idx = t + i * 256;
      int r = idx >> 3, c = (idx & 7) << 2;
      float4 v = *reinterpret_cast<const float4*>(A + (size_t)(m0 + r) * D_ + k0 + c);
      bf16* d = &sA[r][c];
      d[0] = (bf16)v.x; d[1] = (bf16)v.y; d[2] = (bf16)v.z; d[3] = (bf16)v.w;
    }
    // stage W tile 128x32 fp32 -> bf16 (4 float4 per thread)
    #pragma unroll
    for (int i = 0; i < 4; ++i) {
      int idx = t + i * 256;
      int r = idx >> 3, c = (idx & 7) << 2;
      float4 v = *reinterpret_cast<const float4*>(W + (size_t)(n0 + r) * D_ + k0 + c);
      bf16* d = &sB[r][c];
      d[0] = (bf16)v.x; d[1] = (bf16)v.y; d[2] = (bf16)v.z; d[3] = (bf16)v.w;
    }
    __syncthreads();

    v16bf af[2];
    #pragma unroll
    for (int sm = 0; sm < 2; ++sm)
      af[sm] = load_a_frag(&sA[mw * 32 + sm * 16 + l16][0], lh);
    #pragma unroll
    for (int nt = 0; nt < 4; ++nt) {
      v16bf bfg = load_b_frag(&sB[nw * 64 + nt * 16 + l16][0], lh);
      #pragma unroll
      for (int sm = 0; sm < 2; ++sm)
        acc[sm][nt] = wmma_bf16(af[sm], bfg, acc[sm][nt]);
    }
    __syncthreads();
  }

  // store to [B,H,S,DH] bf16
  #pragma unroll
  for (int nt = 0; nt < 4; ++nt) {
    int n = n0 + nw * 64 + nt * 16 + l16;
    float bia = bias[n];
    int hh = n >> 6, dh = n & 63;
    #pragma unroll
    for (int sm = 0; sm < 2; ++sm) {
      #pragma unroll
      for (int r = 0; r < 8; ++r) {
        int m = m0 + mw * 32 + sm * 16 + r + 8 * lh;
        int bb = m >> 11, ss = m & (S_ - 1);
        out[(((size_t)bb * H_ + hh) * S_ + ss) * DH_ + dh] =
            (bf16)(acc[sm][nt][r] + bia);
      }
    }
  }
}

// ---------------------------------------------------------------------------
// Kernel 2: Flash attention over [B,H,S,DH] bf16 Q/K/V -> O [B,S,D] bf16.
// Block = 256 thr = 8 waves; 128 queries per block (16 per wave).
// Q fragments live in registers; K tile staged via async-to-LDS when available;
// V staged transposed through VGPRs.
// ---------------------------------------------------------------------------
__global__ __launch_bounds__(256) void k_flash(
    const bf16* __restrict__ Q, const bf16* __restrict__ K,
    const bf16* __restrict__ V, bf16* __restrict__ O)
{
  __shared__ __align__(16) bf16 sK[64][72];      // [kv][dh]
  __shared__ __align__(16) bf16 sVT[64][72];     // [dh][kv]
  __shared__ __align__(16) bf16 sP[8][16][72];   // per-wave P tile [q][kv]

  const int t = threadIdx.x;
  const int wave = t >> 5, lane = t & 31;
  const int lh = lane >> 4, l16 = lane & 15;
  const int bh = blockIdx.y;
  const size_t base = (size_t)bh * S_ * DH_;
  const int q0 = blockIdx.x * 128 + wave * 16;

  // Q fragments (16 rows x 64 dh) kept in registers for the whole kernel
  const bf16* qrow = Q + base + (size_t)(q0 + l16) * DH_;
  v16bf qf[2];
  qf[0] = load_a_frag(qrow, lh);
  qf[1] = load_a_frag(qrow + 32, lh);

  v8f oacc[4] = { V8F_ZERO, V8F_ZERO, V8F_ZERO, V8F_ZERO };
  float mrow[8], lrow[8];
  #pragma unroll
  for (int r = 0; r < 8; ++r) { mrow[r] = -1e30f; lrow[r] = 0.f; }

  for (int kt = 0; kt < S_ / 64; ++kt) {
    const bf16* kb = K + base + (size_t)kt * 64 * DH_;
    const bf16* vb = V + base + (size_t)kt * 64 * DH_;
    // stage K tile [64][64] bf16 (512 x 16B chunks, 2 per thread)
#if HAVE_ASYNC_LDS
    #pragma unroll
    for (int i = 0; i < 2; ++i) {
      int idx = t + i * 256;
      int r = idx >> 3, c = (idx & 7) << 3;
      ASYNC_CP16(kb + r * 64 + c, &sK[r][c]);
    }
#else
    #pragma unroll
    for (int i = 0; i < 2; ++i) {
      int idx = t + i * 256;
      int r = idx >> 3, c = (idx & 7) << 3;
      *reinterpret_cast<uint4*>(&sK[r][c]) =
          *reinterpret_cast<const uint4*>(kb + r * 64 + c);
    }
#endif
    // stage V transposed: sVT[dh][kv]
    #pragma unroll
    for (int i = 0; i < 8; ++i) {
      int idx = t + i * 256;
      int kv = idx >> 5, dh = (idx & 31) << 1;
      bf16 a0 = vb[kv * 64 + dh];
      bf16 a1 = vb[kv * 64 + dh + 1];
      sVT[dh][kv] = a0;
      sVT[dh + 1][kv] = a1;
    }
    if (kt + 1 < S_ / 64) __builtin_prefetch(kb + 64 * DH_, 0, 1);
#if HAVE_ASYNC_LDS
    __builtin_amdgcn_s_wait_asynccnt(0);
#endif
    __syncthreads();

    // S = Q K^T  (16q x 64kv per wave)
    v8f sacc[4] = { V8F_ZERO, V8F_ZERO, V8F_ZERO, V8F_ZERO };
    #pragma unroll
    for (int nt = 0; nt < 4; ++nt) {
      #pragma unroll
      for (int ks = 0; ks < 2; ++ks) {
        v16bf bfg = load_b_frag(&sK[nt * 16 + l16][ks * 32], lh);
        sacc[nt] = wmma_bf16(qf[ks], bfg, sacc[nt]);
      }
    }
    // scale 1/sqrt(DH)
    #pragma unroll
    for (int nt = 0; nt < 4; ++nt)
      #pragma unroll
      for (int r = 0; r < 8; ++r)
        sacc[nt][r] *= 0.125f;

    // online softmax: row m = r + 8*lh lives in one 16-lane half at vgpr r
    float rm[8];
    #pragma unroll
    for (int r = 0; r < 8; ++r) {
      float v0 = fmaxf(fmaxf(sacc[0][r], sacc[1][r]),
                       fmaxf(sacc[2][r], sacc[3][r]));
      #pragma unroll
      for (int msk = 1; msk < 16; msk <<= 1)
        v0 = fmaxf(v0, __shfl_xor(v0, msk, 32));
      rm[r] = v0;
    }
    #pragma unroll
    for (int r = 0; r < 8; ++r) {
      float mn = fmaxf(mrow[r], rm[r]);
      float al = __expf(mrow[r] - mn);
      mrow[r] = mn;
      lrow[r] *= al;
      #pragma unroll
      for (int nt = 0; nt < 4; ++nt) oacc[nt][r] *= al;
      float rs = 0.f;
      #pragma unroll
      for (int nt = 0; nt < 4; ++nt) {
        float p = __expf(sacc[nt][r] - mn);
        sacc[nt][r] = p;
        rs += p;
      }
      #pragma unroll
      for (int msk = 1; msk < 16; msk <<= 1)
        rs += __shfl_xor(rs, msk, 32);
      lrow[r] += rs;
    }

    // write P (bf16) to this wave's LDS tile
    #pragma unroll
    for (int nt = 0; nt < 4; ++nt)
      #pragma unroll
      for (int r = 0; r < 8; ++r)
        sP[wave][r + 8 * lh][nt * 16 + l16] = (bf16)sacc[nt][r];
    __syncthreads();

    // O += P * V
    v16bf pf[2];
    pf[0] = load_a_frag(&sP[wave][l16][0], lh);
    pf[1] = load_a_frag(&sP[wave][l16][32], lh);
    #pragma unroll
    for (int nt = 0; nt < 4; ++nt) {
      #pragma unroll
      for (int ks = 0; ks < 2; ++ks) {
        v16bf vf = load_b_frag(&sVT[nt * 16 + l16][ks * 32], lh);
        oacc[nt] = wmma_bf16(pf[ks], vf, oacc[nt]);
      }
    }
    __syncthreads();
  }

  // finalize: O /= l ; write to [B,S,D] with D index = h*DH + dh
  const int bb = bh / H_, hh = bh % H_;
  bf16* obase = O + (size_t)bb * S_ * D_ + (size_t)hh * DH_;
  float inv[8];
  #pragma unroll
  for (int r = 0; r < 8; ++r) inv[r] = 1.0f / lrow[r];
  #pragma unroll
  for (int nt = 0; nt < 4; ++nt) {
    int dh = nt * 16 + l16;
    #pragma unroll
    for (int r = 0; r < 8; ++r) {
      int q = q0 + r + 8 * lh;
      obase[(size_t)q * D_ + dh] = (bf16)(oacc[nt][r] * inv[r]);
    }
  }
}

// ---------------------------------------------------------------------------
// Kernel 3: output projection. out[m,n] = sum_k O[m,k]*Wo[n,k] + bo[n] (fp32 out)
// Block tile 128x128, per-wave 32x64 (8 WMMA/K-step). A staged async when possible.
// ---------------------------------------------------------------------------
__global__ __launch_bounds__(256) void k_out_proj(
    const bf16* __restrict__ A, const float* __restrict__ W,
    const float* __restrict__ bias, float* __restrict__ out)
{
  __shared__ __align__(16) bf16 sA[128][48];
  __shared__ __align__(16) bf16 sB[128][48];

  const int t = threadIdx.x;
  const int wave = t >> 5, lane = t & 31;
  const int lh = lane >> 4, l16 = lane & 15;
  const int mw = wave & 3, nw = wave >> 2;
  const int m0 = blockIdx.x * 128;
  const int n0 = blockIdx.y * 128;

  v8f acc[2][4] = { { V8F_ZERO, V8F_ZERO, V8F_ZERO, V8F_ZERO },
                    { V8F_ZERO, V8F_ZERO, V8F_ZERO, V8F_ZERO } };

  for (int k0 = 0; k0 < D_; k0 += 32) {
    // stage A tile 128x32 bf16 (512 x 16B chunks, 2 per thread)
#if HAVE_ASYNC_LDS
    #pragma unroll
    for (int i = 0; i < 2; ++i) {
      int idx = t + i * 256;
      int r = idx >> 2, c = (idx & 3) << 3;
      ASYNC_CP16(A + (size_t)(m0 + r) * D_ + k0 + c, &sA[r][c]);
    }
#else
    #pragma unroll
    for (int i = 0; i < 2; ++i) {
      int idx = t + i * 256;
      int r = idx >> 2, c = (idx & 3) << 3;
      *reinterpret_cast<uint4*>(&sA[r][c]) =
          *reinterpret_cast<const uint4*>(A + (size_t)(m0 + r) * D_ + k0 + c);
    }
#endif
    // stage W tile 128x32 fp32 -> bf16
    #pragma unroll
    for (int i = 0; i < 4; ++i) {
      int idx = t + i * 256;
      int r = idx >> 3, c = (idx & 7) << 2;
      float4 v = *reinterpret_cast<const float4*>(W + (size_t)(n0 + r) * D_ + k0 + c);
      bf16* d = &sB[r][c];
      d[0] = (bf16)v.x; d[1] = (bf16)v.y; d[2] = (bf16)v.z; d[3] = (bf16)v.w;
    }
#if HAVE_ASYNC_LDS
    __builtin_amdgcn_s_wait_asynccnt(0);
#endif
    __syncthreads();

    v16bf af[2];
    #pragma unroll
    for (int sm = 0; sm < 2; ++sm)
      af[sm] = load_a_frag(&sA[mw * 32 + sm * 16 + l16][0], lh);
    #pragma unroll
    for (int nt = 0; nt < 4; ++nt) {
      v16bf bfg = load_b_frag(&sB[nw * 64 + nt * 16 + l16][0], lh);
      #pragma unroll
      for (int sm = 0; sm < 2; ++sm)
        acc[sm][nt] = wmma_bf16(af[sm], bfg, acc[sm][nt]);
    }
    __syncthreads();
  }

  #pragma unroll
  for (int nt = 0; nt < 4; ++nt) {
    int n = n0 + nw * 64 + nt * 16 + l16;
    float bia = bias[n];
    #pragma unroll
    for (int sm = 0; sm < 2; ++sm) {
      #pragma unroll
      for (int r = 0; r < 8; ++r) {
        int m = m0 + mw * 32 + sm * 16 + r + 8 * lh;
        out[(size_t)m * D_ + n] = acc[sm][nt][r] + bia;
      }
    }
  }
}

extern "C" void kernel_launch(void* const* d_in, const int* in_sizes, int n_in,
                              void* d_out, int out_size, void* d_ws, size_t ws_size,
                              hipStream_t stream) {
  (void)in_sizes; (void)n_in; (void)out_size; (void)ws_size;
  const float* xq  = (const float*)d_in[0];
  const float* xkv = (const float*)d_in[1];
  const float* Wq  = (const float*)d_in[2];
  const float* bq  = (const float*)d_in[3];
  const float* Wk  = (const float*)d_in[4];
  const float* bk  = (const float*)d_in[5];
  const float* Wv  = (const float*)d_in[6];
  const float* bv  = (const float*)d_in[7];
  const float* Wo  = (const float*)d_in[8];
  const float* bo  = (const float*)d_in[9];

  const size_t nQ = (size_t)B_ * H_ * S_ * DH_;   // 8,388,608 elems (16 MB bf16)
  bf16* Qw = (bf16*)d_ws;
  bf16* Kw = Qw + nQ;
  bf16* Vw = Kw + nQ;
  bf16* Ow = Vw + nQ;   // [B,S,D] bf16, 16 MB   (total ws use: 64 MB)

  dim3 g1(M_ / 128, D_ / 128, 3);
  k_qkv_proj<<<g1, 256, 0, stream>>>(xq, xkv, Wq, bq, Wk, bk, Wv, bv, Qw, Kw, Vw);

  dim3 g2(S_ / 128, B_ * H_);
  k_flash<<<g2, 256, 0, stream>>>(Qw, Kw, Vw, Ow);

  dim3 g3(M_ / 128, D_ / 128);
  k_out_proj<<<g3, 256, 0, stream>>>(Ow, Wo, bo, (float*)d_out);
}
